// CustomGNN_26285199851521
// MI455X (gfx1250) — matrix operand order repaired and checked by reading, not compiled, not run on hardware
//
#include <hip/hip_runtime.h>

#define NN 50000
#define NE 600000
#define HID 128
#define NG 512
#define WT_LD 132   // padded f16 stride: lane stride 66 dwords -> bank-conflict free

typedef __attribute__((ext_vector_type(16))) _Float16 v16h;
typedef __attribute__((ext_vector_type(8)))  float    v8f;

// ---------------- utility kernels ----------------

__global__ void zero_kernel(float* __restrict__ p, int n) {
  int i = blockIdx.x * blockDim.x + threadIdx.x;
  if (i < n) p[i] = 0.0f;
}

// one wave (32 lanes) per edge; float4 per lane -> 4 atomic f32 adds
__global__ void scatter_add_kernel(const int* __restrict__ src, const int* __restrict__ dst,
                                   const float* __restrict__ feat, float* __restrict__ acc,
                                   int nedges) {
  int gid  = blockIdx.x * blockDim.x + threadIdx.x;
  int e    = gid >> 5;
  int lane = gid & 31;
  if (e >= nedges) return;
  int s = src[e], d = dst[e];
  float4 v = *(const float4*)(feat + (size_t)s * HID + lane * 4);
  float* o = acc + (size_t)d * HID + lane * 4;
  atomicAdd(o + 0, v.x);
  atomicAdd(o + 1, v.y);
  atomicAdd(o + 2, v.z);
  atomicAdd(o + 3, v.w);
}

__global__ void degree_kernel(const int* __restrict__ dst, float* __restrict__ deg, int nedges) {
  int i = blockIdx.x * blockDim.x + threadIdx.x;
  if (i < nedges) atomicAdd(&deg[dst[i]], 1.0f);
}

__global__ void invdeg_kernel(float* __restrict__ deg, int n) {
  int i = blockIdx.x * blockDim.x + threadIdx.x;
  if (i < n) deg[i] = 1.0f / fmaxf(deg[i], 1.0f);
}

// ---------------- WMMA GEMM: out[n x 128] = op(A (+Aadd)) * W + bias ----------------
// A fragments (f16 16x32): lane L -> row M = L&15; K groups of 8 chosen by L>>4.
// B fragments (f16 32x16): lane L -> col N = L&15; K groups of 8 chosen by L>>4, same order as A.
// C (f32 16x16): elem i -> M = i + 8*(L>>4), N = L&15.
__global__ __launch_bounds__(256) void gemm128_wmma_kernel(
    const float* __restrict__ A, const float* __restrict__ Aadd,
    const float* __restrict__ rowscale,
    const float* __restrict__ W, const float* __restrict__ bias,
    float* __restrict__ out, int nrows, int do_relu, int do_acc)
{
  __shared__ _Float16 wt[HID * WT_LD];   // wt[n][k] = W[k][n], f16, padded stride

  int tid = threadIdx.x;
  for (int idx = tid; idx < HID * HID; idx += 256) {
    int n = idx & (HID - 1);
    int k = idx >> 7;
    wt[n * WT_LD + k] = (_Float16)W[k * HID + n];
  }
  __syncthreads();

  int wave = tid >> 5;
  int lane = tid & 31;
  int row0 = (blockIdx.x * 8 + wave) * 16;
  if (row0 >= nrows) return;           // whole-wave exit: EXEC stays all-ones for WMMA

  int m   = lane & 15;
  int hi  = lane >> 4;                 // 0 or 1
  int row = row0 + m;
  const float* rowA = A + (size_t)row * HID;
  const float* rowB = Aadd ? (Aadd + (size_t)row * HID) : nullptr;
  float scale = rowscale ? rowscale[row] : 1.0f;

  v8f acc[8] = {};

#pragma unroll
  for (int ks = 0; ks < 4; ++ks) {
    int k0 = ks * 32;
    int kg = k0 + hi * 8;

    float tmp[16];
    *(float4*)(tmp + 0)  = *(const float4*)(rowA + kg);
    *(float4*)(tmp + 4)  = *(const float4*)(rowA + kg + 4);
    *(float4*)(tmp + 8)  = *(const float4*)(rowA + kg + 16);
    *(float4*)(tmp + 12) = *(const float4*)(rowA + kg + 20);
    if (rowB) {
      float t2[16];
      *(float4*)(t2 + 0)  = *(const float4*)(rowB + kg);
      *(float4*)(t2 + 4)  = *(const float4*)(rowB + kg + 4);
      *(float4*)(t2 + 8)  = *(const float4*)(rowB + kg + 16);
      *(float4*)(t2 + 12) = *(const float4*)(rowB + kg + 20);
#pragma unroll
      for (int j = 0; j < 16; ++j) tmp[j] += t2[j];
    }
    v16h afrag;
#pragma unroll
    for (int j = 0; j < 16; ++j) afrag[j] = (_Float16)(tmp[j] * scale);

#pragma unroll
    for (int t = 0; t < 8; ++t) {
      int n = t * 16 + m;
      const unsigned int* pw = (const unsigned int*)(wt + n * WT_LD + k0 + hi * 8);
      union { v16h h; unsigned int u[8]; } bf;
#pragma unroll
      for (int j = 0; j < 8; ++j) bf.u[j] = pw[(j & 3) + (j >> 2) * 8];
      acc[t] = __builtin_amdgcn_wmma_f32_16x16x32_f16(
          false, afrag, false, bf.h, (short)0, acc[t], false, false);
    }
  }

#pragma unroll
  for (int t = 0; t < 8; ++t) {
    int c = t * 16 + m;
    float bv = bias ? bias[c] : 0.0f;
#pragma unroll
    for (int i = 0; i < 8; ++i) {
      int r = row0 + hi * 8 + i;
      size_t o = (size_t)r * HID + c;
      float v = acc[t][i] + bv;
      if (do_acc)  v += out[o];
      if (do_relu) v  = fmaxf(v, 0.0f);
      out[o] = v;
    }
  }
}

// ---------------- BatchNorm (training-mode, biased var) ----------------

__global__ void bn_stats_kernel(const float* __restrict__ h, float* __restrict__ stats, int n) {
  // stats[0..127] = sum, stats[128..255] = sumsq
  int t = blockIdx.x * blockDim.x + threadIdx.x;
  int f = t & (HID - 1);
  int chunk = t >> 7;
  int nchunks = (gridDim.x * blockDim.x) >> 7;
  float s = 0.0f, ss = 0.0f;
  for (int i = chunk; i < n; i += nchunks) {
    float v = h[(size_t)i * HID + f];
    s += v; ss += v * v;
  }
  atomicAdd(&stats[f], s);
  atomicAdd(&stats[HID + f], ss);
}

__global__ void bn_apply_kernel(const float* __restrict__ pre, const float* __restrict__ stats,
                                const float* __restrict__ gamma, const float* __restrict__ beta,
                                float* __restrict__ out, int n) {
  int i = blockIdx.x * blockDim.x + threadIdx.x;
  if (i >= n * HID) return;
  int f = i & (HID - 1);
  float inv_n = 1.0f / (float)n;
  float mean = stats[f] * inv_n;
  float var  = stats[HID + f] * inv_n - mean * mean;
  float sc = gamma[f] * rsqrtf(var + 1e-5f);
  float sh = beta[f] - sc * mean;
  out[i] = fmaxf(pre[i] * sc + sh, 0.0f);
}

// ---------------- global mean pool ----------------

__global__ void pool_kernel(const float* __restrict__ h, const int* __restrict__ batch,
                            float* __restrict__ sums, float* __restrict__ counts, int n) {
  int gid  = blockIdx.x * blockDim.x + threadIdx.x;
  int node = gid >> 5;
  int lane = gid & 31;
  if (node >= n) return;
  int b = batch[node];
  float4 v = *(const float4*)(h + (size_t)node * HID + lane * 4);
  float* o = sums + (size_t)b * HID + lane * 4;
  atomicAdd(o + 0, v.x);
  atomicAdd(o + 1, v.y);
  atomicAdd(o + 2, v.z);
  atomicAdd(o + 3, v.w);
  if (lane == 0) atomicAdd(&counts[b], 1.0f);
}

__global__ void pool_finalize_kernel(const float* __restrict__ sums, const float* __restrict__ counts,
                                     float* __restrict__ out) {
  int i = blockIdx.x * blockDim.x + threadIdx.x;
  if (i >= NG * HID) return;
  int g = i >> 7;
  out[i] = sums[i] / fmaxf(counts[g], 1.0f);
}

// ---------------- driver ----------------

static inline int cdiv(int a, int b) { return (a + b - 1) / b; }

extern "C" void kernel_launch(void* const* d_in, const int* in_sizes, int n_in,
                              void* d_out, int out_size, void* d_ws, size_t ws_size,
                              hipStream_t stream) {
  const float* x       = (const float*)d_in[0];
  const int*   eidx    = (const int*)d_in[1];
  const int*   src     = eidx;
  const int*   dst     = eidx + NE;
  const int*   batch   = (const int*)d_in[2];
  const float* gin_w1  = (const float*)d_in[3];
  const float* gin_b1  = (const float*)d_in[4];
  const float* gin_w2  = (const float*)d_in[5];
  const float* gin_b2  = (const float*)d_in[6];
  const float* sage_wl = (const float*)d_in[7];
  const float* sage_bl = (const float*)d_in[8];
  const float* sage_wr = (const float*)d_in[9];
  const float* bn_g    = (const float*)d_in[10];
  const float* bn_b    = (const float*)d_in[11];
  float* outp = (float*)d_out;

  // workspace layout (f32)
  char* ws = (char*)d_ws;
  float* buf0   = (float*)(ws);                                   // 50000*128
  float* buf1   = (float*)(ws + (size_t)NN * HID * 4);            // 50000*128
  float* deg    = (float*)(ws + (size_t)2 * NN * HID * 4);        // 50000
  float* stats  = (float*)(ws + (size_t)2 * NN * HID * 4 + 256 * 1024);       // 256
  float* sums   = (float*)(ws + (size_t)2 * NN * HID * 4 + 256 * 1024 + 4096);// 512*128
  float* counts = sums + (size_t)NG * HID;                        // 512

  const int feat_elems = NN * HID;
  dim3 blk(256);

  // degree (shared by SAGE layers)
  zero_kernel<<<cdiv(NN, 256), blk, 0, stream>>>(deg, NN);
  degree_kernel<<<cdiv(NE, 256), blk, 0, stream>>>(dst, deg, NE);
  invdeg_kernel<<<cdiv(NN, 256), blk, 0, stream>>>(deg, NN);

  // ---- GIN: agg = segsum(x[src], dst); h = mlp(x + agg) ----
  zero_kernel<<<cdiv(feat_elems, 256), blk, 0, stream>>>(buf0, feat_elems);
  scatter_add_kernel<<<cdiv(NE * 32, 256), blk, 0, stream>>>(src, dst, x, buf0, NE);

  int gemm_grid = cdiv(NN / 16, 8);  // 3125 strips / 8 waves
  // pass1: buf1 = relu((x + buf0) @ W1 + b1)
  gemm128_wmma_kernel<<<gemm_grid, blk, 0, stream>>>(buf0, x, nullptr, gin_w1, gin_b1,
                                                     buf1, NN, 1, 0);
  // pass2: buf0 = buf1 @ W2 + b2
  gemm128_wmma_kernel<<<gemm_grid, blk, 0, stream>>>(buf1, nullptr, nullptr, gin_w2, gin_b2,
                                                     buf0, NN, 0, 0);
  // BN + relu -> buf1 (h)
  zero_kernel<<<1, blk, 0, stream>>>(stats, 2 * HID);
  bn_stats_kernel<<<512, blk, 0, stream>>>(buf0, stats, NN);
  bn_apply_kernel<<<cdiv(feat_elems, 256), blk, 0, stream>>>(buf0, stats, bn_g, bn_b, buf1, NN);

  // ---- SAGE layers ----
  for (int l = 0; l < 2; ++l) {
    const float* wl = sage_wl + (size_t)l * HID * HID;
    const float* bl = sage_bl + (size_t)l * HID;
    const float* wr = sage_wr + (size_t)l * HID * HID;
    zero_kernel<<<cdiv(feat_elems, 256), blk, 0, stream>>>(buf0, feat_elems);
    scatter_add_kernel<<<cdiv(NE * 32, 256), blk, 0, stream>>>(src, dst, buf1, buf0, NE);
    // buf0 = (buf0 * invdeg) @ Wl + bl      (in-place per-strip safe)
    gemm128_wmma_kernel<<<gemm_grid, blk, 0, stream>>>(buf0, nullptr, deg, wl, bl,
                                                       buf0, NN, 0, 0);
    // buf0 += buf1 @ Wr
    gemm128_wmma_kernel<<<gemm_grid, blk, 0, stream>>>(buf1, nullptr, nullptr, wr, nullptr,
                                                       buf0, NN, 0, 1);
    // BN + relu -> buf1
    zero_kernel<<<1, blk, 0, stream>>>(stats, 2 * HID);
    bn_stats_kernel<<<512, blk, 0, stream>>>(buf0, stats, NN);
    bn_apply_kernel<<<cdiv(feat_elems, 256), blk, 0, stream>>>(buf0, stats,
                                                               bn_g + (size_t)(l + 1) * HID,
                                                               bn_b + (size_t)(l + 1) * HID,
                                                               buf1, NN);
  }

  // ---- global mean pool ----
  zero_kernel<<<cdiv(NG * HID + NG, 256), blk, 0, stream>>>(sums, NG * HID + NG);
  pool_kernel<<<cdiv(NN * 32, 256), blk, 0, stream>>>(buf1, batch, sums, counts, NN);
  pool_finalize_kernel<<<cdiv(NG * HID, 256), blk, 0, stream>>>(sums, counts, outp);
}
